// LSTM_2723009266181
// MI455X (gfx1250) — compile-verified, matrix operand order checked by reading, not run on hardware
//
#include <hip/hip_runtime.h>

// Masked LSTM, L=512, N=512, I=H=128 — fused single-kernel recurrence.
// gates = [h|x] @ [W_hh|W_ih]^T via v_wmma_f32_16x16x32_bf16 (f32 accumulate).
// 32 persistent workgroups (one 16-batch-row tile each), 1024 threads = 32 waves,
// wave w owns gate columns [16w,16w+16); its 8 B fragments live in VGPRs for all
// 512 steps. h/x staged in LDS as bf16 (direct WMMA A-fragment loads, no cvt),
// c kept f32. x double-buffered with a global prefetch issued one step ahead.

#define LL 512
#define NN 512
#define II 128
#define HH 128

typedef __bf16 v16bf __attribute__((ext_vector_type(16)));
typedef __bf16 bf8v  __attribute__((ext_vector_type(8)));
typedef __bf16 bf2v  __attribute__((ext_vector_type(2)));
typedef float  v8f   __attribute__((ext_vector_type(8)));

#define SPH  136   // bf16 row stride for h/x tiles: 272 B -> rows hit distinct banks
#define SPC  132   // f32 row stride for c tile
#define GSTR 516   // f32 row stride for gates tile (8*516 % 64 != 0)

__device__ __forceinline__ float sigm(float x)   { return 1.0f / (1.0f + __expf(-x)); }
__device__ __forceinline__ float tanh_f(float x) { return 1.0f - 2.0f / (1.0f + __expf(2.0f * x)); }

__launch_bounds__(1024, 1)
__global__ void lstm_fused_kernel(const float* __restrict__ input,
                                  const float* __restrict__ hc,
                                  const int*   __restrict__ is_initial,
                                  const float* __restrict__ W_ih,
                                  const float* __restrict__ W_hh,
                                  const float* __restrict__ b_ih,
                                  const float* __restrict__ b_hh,
                                  float* __restrict__ out)
{
    __shared__ __align__(16) __bf16 h_bf[16 * SPH];
    __shared__ __align__(16) __bf16 x_bf[2 * 16 * SPH];   // double buffered
    __shared__ __align__(16) float  c_lds[16 * SPC];
    __shared__ __align__(16) float  g_lds[16 * GSTR];
    __shared__ __align__(16) float  bias_lds[4 * HH];

    const int tid  = threadIdx.x;
    const int lane = tid & 31;
    const int w    = tid >> 5;            // wave id 0..31 == gate-column tile
    const int n0   = blockIdx.x * 16;     // batch-row base for this block

    // elementwise mapping: thread -> (row m, even column hx0), one bf16/f32 pair
    const int em  = tid >> 6;             // 0..15
    const int ex0 = (tid & 63) * 2;       // 0,2,..,126

    // ---- init: combined bias, h0/c0 from hc, x_0 into buffer 0 ----
    for (int q = tid; q < 4 * HH; q += 1024)
        bias_lds[q] = b_ih[q] + b_hh[q];
    {
        const float2 h2 = *(const float2*)(hc + (size_t)(n0 + em) * 256 + ex0);
        const float2 c2 = *(const float2*)(hc + (size_t)(n0 + em) * 256 + 128 + ex0);
        bf2v hb; hb[0] = (__bf16)h2.x; hb[1] = (__bf16)h2.y;
        *(bf2v*)(h_bf + em * SPH + ex0) = hb;
        c_lds[em * SPC + ex0]     = c2.x;
        c_lds[em * SPC + ex0 + 1] = c2.y;
        const float2 x2 = *(const float2*)(input + ((size_t)0 * NN + n0 + em) * II + ex0);
        bf2v xb; xb[0] = (__bf16)x2.x; xb[1] = (__bf16)x2.y;
        *(bf2v*)(x_bf + em * SPH + ex0) = xb;
    }

    // ---- constant B fragments: B[k][j] = k<128 ? W_hh[j][k] : W_ih[j][k-128]
    // 16-bit B 32x16 layout: lane holds column n=lane&15; K runs {k0+base..+7, k0+16+base..+7},
    // base = 0 (lanes 0-15) / 8 (lanes 16-31).
    const int n    = lane & 15;
    const int base = (lane & 16) ? 8 : 0;
    const int j    = w * 16 + n;          // gate column 0..511
    v16bf B[8];
#pragma unroll
    for (int kf = 0; kf < 8; ++kf) {
        int k0 = kf * 32;
        v16bf b;
#pragma unroll
        for (int run = 0; run < 2; ++run) {
            int k = k0 + run * 16 + base;   // run of 8 contiguous K, never crosses the 128 split
            const float* src = (k < 128) ? (W_hh + (size_t)j * 128 + k)
                                         : (W_ih + (size_t)j * 128 + (k - 128));
#pragma unroll
            for (int e = 0; e < 8; ++e)
                b[run * 8 + e] = (__bf16)src[e];
        }
        B[kf] = b;
    }
    __syncthreads();

    const int moff = (lane & 16) ? 8 : 0;   // C-tile: M = vgpr + moff, N = lane&15
    const int mA   = lane & 15;             // A-tile: M = lane&15

    for (int t = 0; t < LL; ++t) {
        // ---- phase 1: episode-reset mask (exact select, is_initial in {0,1});
        //               issue global prefetch of x_{t+1} (consumed in phase 3) ----
        float2 xnext = make_float2(0.0f, 0.0f);
        {
            const int ini = is_initial[(size_t)t * NN + n0 + em];
            unsigned int* hp = (unsigned int*)(h_bf + em * SPH + ex0);
            unsigned int  hv = *hp;
            *hp = ini ? 0u : hv;
            float* cp = c_lds + em * SPC + ex0;
            float c0 = cp[0], c1 = cp[1];
            cp[0] = ini ? 0.0f : c0;
            cp[1] = ini ? 0.0f : c1;
            if (t + 1 < LL)
                xnext = *(const float2*)(input + ((size_t)(t + 1) * NN + n0 + em) * II + ex0);
        }
        __syncthreads();

        // ---- phase 2: gates tile = [h|x] @ B  (8 x v_wmma_f32_16x16x32_bf16) ----
        {
            const __bf16* xcur = x_bf + (t & 1) * 16 * SPH;
            v16bf a[8];
#pragma unroll
            for (int kf = 0; kf < 8; ++kf) {
                const __bf16* src = (kf < 4) ? (h_bf + mA * SPH + kf * 32 + base)
                                             : (xcur + mA * SPH + (kf - 4) * 32 + base);
                bf8v lo = *(const bf8v*)(src);        // K = k0+base   .. +7
                bf8v hi = *(const bf8v*)(src + 16);   // K = k0+16+base.. +7
                a[kf] = __builtin_shufflevector(lo, hi, 0, 1, 2, 3, 4, 5, 6, 7,
                                                        8, 9, 10, 11, 12, 13, 14, 15);
            }
            v8f acc = {};
#pragma unroll
            for (int kf = 0; kf < 8; ++kf)
                acc = __builtin_amdgcn_wmma_f32_16x16x32_bf16(
                    false, a[kf], false, B[kf], (short)0, acc, false, false);
#pragma unroll
            for (int r = 0; r < 8; ++r)
                g_lds[(r + moff) * GSTR + w * 16 + n] = acc[r];
        }
        __syncthreads();

        // ---- phase 3: activations, state update, stream h out, park prefetched x ----
        {
            const float* gr = g_lds + em * GSTR;
            const float2 gi = *(const float2*)(gr +        ex0);
            const float2 gf = *(const float2*)(gr + 128 + ex0);
            const float2 gg = *(const float2*)(gr + 256 + ex0);
            const float2 go = *(const float2*)(gr + 384 + ex0);
            float cn0, cn1, hn0, hn1;
            {
                float ig = sigm  (gi.x + bias_lds[        ex0]);
                float fg = sigm  (gf.x + bias_lds[128   + ex0]);
                float g_ = tanh_f(gg.x + bias_lds[256   + ex0]);
                float og = sigm  (go.x + bias_lds[384   + ex0]);
                cn0 = fg * c_lds[em * SPC + ex0] + ig * g_;
                hn0 = og * tanh_f(cn0);
            }
            {
                float ig = sigm  (gi.y + bias_lds[        ex0 + 1]);
                float fg = sigm  (gf.y + bias_lds[128   + ex0 + 1]);
                float g_ = tanh_f(gg.y + bias_lds[256   + ex0 + 1]);
                float og = sigm  (go.y + bias_lds[384   + ex0 + 1]);
                cn1 = fg * c_lds[em * SPC + ex0 + 1] + ig * g_;
                hn1 = og * tanh_f(cn1);
            }
            c_lds[em * SPC + ex0]     = cn0;
            c_lds[em * SPC + ex0 + 1] = cn1;
            bf2v hb; hb[0] = (__bf16)hn0; hb[1] = (__bf16)hn1;
            *(bf2v*)(h_bf + em * SPH + ex0) = hb;
            *(float2*)(out + ((size_t)t * NN + n0 + em) * HH + ex0) = make_float2(hn0, hn1);

            if (t == LL - 1) {   // final hc = concat(h, c), full f32 from registers
                float* hcout = out + (size_t)LL * NN * HH;
                *(float2*)(hcout + (size_t)(n0 + em) * 256 + ex0)       = make_float2(hn0, hn1);
                *(float2*)(hcout + (size_t)(n0 + em) * 256 + 128 + ex0) = make_float2(cn0, cn1);
            }

            // park x_{t+1} into the alternate buffer (zeros past the end, unused)
            bf2v xb; xb[0] = (__bf16)xnext.x; xb[1] = (__bf16)xnext.y;
            *(bf2v*)(x_bf + ((t + 1) & 1) * 16 * SPH + em * SPH + ex0) = xb;
        }
        __syncthreads();
    }
}

extern "C" void kernel_launch(void* const* d_in, const int* in_sizes, int n_in,
                              void* d_out, int out_size, void* d_ws, size_t ws_size,
                              hipStream_t stream)
{
    (void)in_sizes; (void)n_in; (void)out_size; (void)d_ws; (void)ws_size;
    const float* input      = (const float*)d_in[0];
    const float* hc         = (const float*)d_in[1];
    const int*   is_initial = (const int*)  d_in[2];
    const float* W_ih       = (const float*)d_in[3];
    const float* W_hh       = (const float*)d_in[4];
    const float* b_ih       = (const float*)d_in[5];
    const float* b_hh       = (const float*)d_in[6];
    float* out = (float*)d_out;

    lstm_fused_kernel<<<dim3(NN / 16), dim3(1024), 0, stream>>>(
        input, hc, is_initial, W_ih, W_hh, b_ih, b_hh, out);
}
